// SA_1700807049802
// MI455X (gfx1250) — matrix operand order repaired and checked by reading, not compiled, not run on hardware
//
#include <hip/hip_runtime.h>
#include <cstdint>

// ---------------------------------------------------------------------------
// Transformer encoder layer forward for MI455X (gfx1250, wave32).
// bf16-resident dataflow; GLOBAL_LOAD_ASYNC_TO_LDS_B128 double-buffered
// staging; v_wmma_f32_16x16x32_bf16 everywhere; DPP16 softmax reductions.
// Workspace requirement ~190 MB.
// ---------------------------------------------------------------------------

#define DEVINL __device__ __forceinline__

typedef __attribute__((ext_vector_type(16))) __bf16 v16bf;
typedef __attribute__((ext_vector_type(8)))  float  v8f;

static constexpr int Bc   = 4;
static constexpr int Sc   = 2048;
static constexpr int HIDc = 768;
static constexpr int FFc  = 3072;
static constexpr int NHc  = 12;
static constexpr int HDc  = 64;
static constexpr int Tc   = Bc * Sc;            // 8192 tokens
static constexpr float EPSc = 1e-6f;

DEVINL v8f wmma_bf16(v16bf a, v16bf b, v8f c) {
  return __builtin_amdgcn_wmma_f32_16x16x32_bf16(false, a, false, b,
                                                 (short)0, c, false, false);
}

// Flat LDS pointers carry the byte offset in the low 32 bits.
DEVINL unsigned lds_addr(const void* p) { return (unsigned)(uintptr_t)p; }

// CDNA5 async copy: global -> LDS, 16 bytes per lane, tracked by ASYNCcnt.
DEVINL void async_b128(const void* lds_dst, const void* gsrc) {
  asm volatile("global_load_async_to_lds_b128 %0, %1, off"
               :: "v"(lds_addr(lds_dst)), "v"(gsrc) : "memory");
}
DEVINL void async_wait0() {
#if __has_builtin(__builtin_amdgcn_s_wait_asynccnt)
  __builtin_amdgcn_s_wait_asynccnt(0);
#else
  asm volatile("s_wait_asynccnt 0x0" ::: "memory");
#endif
}

// ---- DPP16 16-lane reductions (VALU only; never cross the 16-lane halves).
template <int CTRL>
DEVINL float dpp_mov(float v) {
  const int s = __builtin_bit_cast(int, v);
  return __builtin_bit_cast(
      float, __builtin_amdgcn_update_dpp(s, s, CTRL, 0xF, 0xF, true));
}
// xor-mask network {1,2,7,15}: masks generate bits 0..3 -> full 16-lane reduce.
DEVINL float red_max16(float v) {
  v = fmaxf(v, dpp_mov<0xB1>(v));    // quad_perm(1,0,3,2)  : xor 1
  v = fmaxf(v, dpp_mov<0x4E>(v));    // quad_perm(2,3,0,1)  : xor 2
  v = fmaxf(v, dpp_mov<0x141>(v));   // row_half_mirror     : xor 7
  v = fmaxf(v, dpp_mov<0x140>(v));   // row_mirror          : xor 15
  return v;
}
DEVINL float red_sum16(float v) {
  v += dpp_mov<0xB1>(v);
  v += dpp_mov<0x4E>(v);
  v += dpp_mov<0x141>(v);
  v += dpp_mov<0x140>(v);
  return v;
}

// A-matrix (16x32, MxK) fragment from row-major bf16 tile (ld elems/row).
// ISA 7.12.2: lanes 0-15 -> M=lane, K={0..7,16..23}; lanes 16-31 -> M=lane-16,
// K={8..15,24..31}; merges into 2x ds_load_b128.
DEVINL v16bf load_a_frag(const __bf16* base, int ld) {
  const int lane = threadIdx.x & 31;
  const int m    = lane & 15;
  const int koff = (lane < 16) ? 0 : 8;
  v16bf f;
#pragma unroll
  for (int i = 0; i < 8; ++i) {
    const int kb = (i < 4) ? (2 * i) : (16 + 2 * (i - 4));
    const int k  = kb + koff;
    f[2 * i]     = base[m * ld + k];
    f[2 * i + 1] = base[m * ld + k + 1];
  }
  return f;
}

// B-matrix (32x16, KxN) fragment, storage [n][k] row-major (weight layout).
DEVINL v16bf load_b_frag_nk(const __bf16* nk, int ld) {
  const int lane = threadIdx.x & 31;
  const int n    = lane & 15;
  const int koff = (lane < 16) ? 0 : 8;
  v16bf f;
#pragma unroll
  for (int i = 0; i < 8; ++i) {
    const int kb = (i < 4) ? (2 * i) : (16 + 2 * (i - 4));
    f[2 * i]     = nk[n * ld + koff + kb];
    f[2 * i + 1] = nk[n * ld + koff + kb + 1];
  }
  return f;
}

// B-matrix (32x16, KxN) fragment, storage [k][n] row-major.
DEVINL v16bf load_b_frag_kn(const __bf16* kn, int ld) {
  const int lane = threadIdx.x & 31;
  const int n    = lane & 15;
  const int koff = (lane < 16) ? 0 : 8;
  v16bf f;
#pragma unroll
  for (int i = 0; i < 8; ++i) {
    const int kb = (i < 4) ? (2 * i) : (16 + 2 * (i - 4));
    f[2 * i]     = kn[(koff + kb) * ld + n];
    f[2 * i + 1] = kn[(koff + kb + 1) * ld + n];
  }
  return f;
}

// ---------------------------------------------------------------------------
// One-time conversions.
// ---------------------------------------------------------------------------
__global__ __launch_bounds__(256)
void cvt_f32_to_bf16(const float* __restrict__ src, __bf16* __restrict__ dst) {
  const size_t i = ((size_t)blockIdx.x * 256 + threadIdx.x) * 4;
  const float4 v = *(const float4*)(src + i);
  dst[i]     = (__bf16)v.x;
  dst[i + 1] = (__bf16)v.y;
  dst[i + 2] = (__bf16)v.z;
  dst[i + 3] = (__bf16)v.w;
}

__global__ __launch_bounds__(256)
void cvt_mask_kernel(const unsigned char* __restrict__ m, float* __restrict__ am) {
  const int i = blockIdx.x * 256 + threadIdx.x;
  am[i] = m[i] ? -1.0e9f : 0.0f;
}

// ---------------------------------------------------------------------------
// Tiled GEMM: Y[T,N] = X[T,K] @ W[N,K]^T + bias (+ReLU). X,W bf16; Y = OutT.
// Block = 256 threads / 8 waves; tile BM=256, BN=64, BK=32; wave w owns a
// 32x64 strip -> 2 A-frags x 4 shared B-frags = 8 WMMAs per K-step
// (1.5 ds_load_b128 per WMMA). Double-buffered async staging, 1 barrier/step.
// ---------------------------------------------------------------------------
template <typename OutT, int N, int K, bool RELU>
__global__ __launch_bounds__(256)
void gemm_bf16_kernel(const __bf16* __restrict__ X, const __bf16* __restrict__ W,
                      const float* __restrict__ bias, OutT* __restrict__ Y) {
  __shared__ __align__(16) __bf16 Xs[2][256 * 32];  // 32 KB
  __shared__ __align__(16) __bf16 Ws[2][64 * 32];   //  8 KB

  const int tid  = threadIdx.x;
  const int wave = tid >> 5;
  const int lane = tid & 31;
  const int t0   = blockIdx.x * 256;
  const int n0   = blockIdx.y * 64;
  constexpr int NT = K / 32;

  // Stage tile `it`: 5 async b128 ops per thread (X: 256x64B, W: 64x64B).
  auto stage = [&](int it, int buf) {
    const int k0 = it * 32;
#pragma unroll
    for (int u = 0; u < 4; ++u) {
      const int idx = tid + u * 256;
      const int r   = idx >> 2;
      const int sg  = (idx & 3) << 3;               // 8 bf16 = 16 B segments
      async_b128(&Xs[buf][r * 32 + sg], X + (size_t)(t0 + r) * K + k0 + sg);
    }
    {
      const int r  = tid >> 2;
      const int sg = (tid & 3) << 3;
      async_b128(&Ws[buf][r * 32 + sg], W + (size_t)(n0 + r) * K + k0 + sg);
    }
  };

  v8f acc0[4], acc1[4];
#pragma unroll
  for (int j = 0; j < 4; ++j)
#pragma unroll
    for (int e = 0; e < 8; ++e) { acc0[j][e] = 0.0f; acc1[j][e] = 0.0f; }

  stage(0, 0);
  for (int it = 0; it < NT; ++it) {
    const int cur = it & 1;
    async_wait0();       // own part of tile `it` landed in LDS
    __syncthreads();     // all waves: tile ready AND tile it-1 fully consumed
    if (it + 1 < NT) stage(it + 1, cur ^ 1);      // overlaps WMMAs below

    const v16bf a0 = load_a_frag(&Xs[cur][(wave * 32) * 32], 32);
    const v16bf a1 = load_a_frag(&Xs[cur][(wave * 32 + 16) * 32], 32);
#pragma unroll
    for (int j = 0; j < 4; ++j) {
      const v16bf b = load_b_frag_nk(&Ws[cur][j * 16 * 32], 32);
      acc0[j] = wmma_bf16(a0, b, acc0[j]);
      acc1[j] = wmma_bf16(a1, b, acc1[j]);
    }
  }

  // Epilogue. C/D layout: lane<16 -> M=i, N=lane; lane>=16 -> M=8+i.
  const int n  = lane & 15;
  const int mb = (lane < 16) ? 0 : 8;
#pragma unroll
  for (int j = 0; j < 4; ++j) {
    const float bv = bias[n0 + j * 16 + n];
#pragma unroll
    for (int i = 0; i < 8; ++i) {
      float v0 = acc0[j][i] + bv;
      float v1 = acc1[j][i] + bv;
      if (RELU) { v0 = fmaxf(v0, 0.0f); v1 = fmaxf(v1, 0.0f); }
      const size_t r0 = (size_t)(t0 + wave * 32 + mb + i);
      Y[r0 * N + n0 + j * 16 + n]        = (OutT)v0;
      Y[(r0 + 16) * N + n0 + j * 16 + n] = (OutT)v1;
    }
  }
}

// ---------------------------------------------------------------------------
// Flash attention: block = (128 q-rows, head, batch), 256 threads / 8 waves.
// q,k,v,out bf16 [B*S, HID], head h at cols [h*64,h*64+64). Double-buffered
// async K/V chunk staging (32 keys), DPP16 online softmax, 1 barrier/chunk.
// ---------------------------------------------------------------------------
__global__ __launch_bounds__(256)
void flash_attn_kernel(const __bf16* __restrict__ q, const __bf16* __restrict__ k,
                       const __bf16* __restrict__ v,
                       const float* __restrict__ amask,   // additive: 0 / -1e9
                       __bf16* __restrict__ out) {
  __shared__ __align__(16) __bf16 Ks[2][32 * 64];   // 8 KB
  __shared__ __align__(16) __bf16 Vs[2][32 * 64];   // 8 KB
  __shared__ __align__(16) __bf16 Ps[8][16 * 32];   // 8 KB per-wave P scratch

  const int tid  = threadIdx.x;
  const int wave = tid >> 5;
  const int lane = tid & 31;
  const int s0   = blockIdx.x * 128;
  const int h    = blockIdx.y;
  const int b    = blockIdx.z;
  const int hcol = h * HDc;
  constexpr int NCHUNK = Sc / 32;

  auto stage = [&](int it, int buf) {
    const int r  = tid >> 3;           // key row 0..31
    const int sg = (tid & 7) << 3;     // 16 B segments of a 64-elem row
    const size_t t = (size_t)(b * Sc + it * 32 + r);
    async_b128(&Ks[buf][r * 64 + sg], k + t * HIDc + hcol + sg);
    async_b128(&Vs[buf][r * 64 + sg], v + t * HIDc + hcol + sg);
  };

  // Q fragments for the two 32-wide K-steps (scale applied to scores later).
  v16bf qa[2];
  {
    const int koff = (lane < 16) ? 0 : 8;
    const __bf16* qp =
        q + (size_t)(b * Sc + s0 + wave * 16 + (lane & 15)) * HIDc + hcol;
#pragma unroll
    for (int ks = 0; ks < 2; ++ks) {
#pragma unroll
      for (int i = 0; i < 8; ++i) {
        const int kb = (i < 4) ? (2 * i) : (16 + 2 * (i - 4));
        const int kk = ks * 32 + kb + koff;
        qa[ks][2 * i]     = qp[kk];
        qa[ks][2 * i + 1] = qp[kk + 1];
      }
    }
  }

  v8f o[4];
#pragma unroll
  for (int j = 0; j < 4; ++j)
#pragma unroll
    for (int e = 0; e < 8; ++e) o[j][e] = 0.0f;
  float mrow[8], lrow[8];
#pragma unroll
  for (int i = 0; i < 8; ++i) { mrow[i] = -3.0e38f; lrow[i] = 0.0f; }

  stage(0, 0);
  for (int it = 0; it < NCHUNK; ++it) {
    const int cur = it & 1;
    async_wait0();
    __syncthreads();     // chunk ready AND chunk it-1 fully consumed
    if (it + 1 < NCHUNK) stage(it + 1, cur ^ 1);

    // Scores S = Q K^T: two 16x16 tiles (keys 0-15 / 16-31 of chunk).
    v8f sc0, sc1;
#pragma unroll
    for (int e = 0; e < 8; ++e) { sc0[e] = 0.0f; sc1[e] = 0.0f; }
#pragma unroll
    for (int ks = 0; ks < 2; ++ks) {
      const v16bf bk0 = load_b_frag_nk(&Ks[cur][ks * 32], 64);
      const v16bf bk1 = load_b_frag_nk(&Ks[cur][16 * 64 + ks * 32], 64);
      sc0 = wmma_bf16(qa[ks], bk0, sc0);
      sc1 = wmma_bf16(qa[ks], bk1, sc1);
    }

    const size_t mbase = (size_t)b * Sc + it * 32;
    const float am0 = amask[mbase + (lane & 15)];
    const float am1 = amask[mbase + 16 + (lane & 15)];

    // Online softmax; DPP16 reductions stay within each 16-lane half.
    float alpha[8];
#pragma unroll
    for (int i = 0; i < 8; ++i) {
      const float a0 = sc0[i] * 0.125f + am0;   // 1/sqrt(64)
      const float a1 = sc1[i] * 0.125f + am1;
      const float mx   = red_max16(fmaxf(a0, a1));
      const float mnew = fmaxf(mrow[i], mx);
      const float al = __expf(mrow[i] - mnew);
      const float p0 = __expf(a0 - mnew);
      const float p1 = __expf(a1 - mnew);
      const float rs = red_sum16(p0 + p1);
      lrow[i] = lrow[i] * al + rs;
      mrow[i] = mnew;
      alpha[i] = al;
      sc0[i] = p0;
      sc1[i] = p1;
    }
#pragma unroll
    for (int j = 0; j < 4; ++j)
#pragma unroll
      for (int i = 0; i < 8; ++i) o[j][i] *= alpha[i];

    // P (D-layout) -> wave-private LDS [16][32] bf16 -> reload as A fragment.
    // DS ops are in-order within a wave; no barrier needed for Ps.
    {
      __bf16* pw = &Ps[wave][0];
      const int n  = lane & 15;
      const int mb = (lane < 16) ? 0 : 8;
#pragma unroll
      for (int i = 0; i < 8; ++i) {
        pw[(mb + i) * 32 + n]      = (__bf16)sc0[i];
        pw[(mb + i) * 32 + 16 + n] = (__bf16)sc1[i];
      }
    }
    const v16bf pa = load_a_frag(&Ps[wave][0], 32);
#pragma unroll
    for (int j = 0; j < 4; ++j) {
      const v16bf bv = load_b_frag_kn(&Vs[cur][j * 16], 64);
      o[j] = wmma_bf16(pa, bv, o[j]);
    }
  }

  // Epilogue: O / l -> bf16 out[t][h*64 + col].
  const int n  = lane & 15;
  const int mb = (lane < 16) ? 0 : 8;
#pragma unroll
  for (int i = 0; i < 8; ++i) {
    const float inv = 1.0f / lrow[i];
    const size_t t = (size_t)(b * Sc + s0 + wave * 16 + mb + i);
#pragma unroll
    for (int j = 0; j < 4; ++j)
      out[t * HIDc + hcol + j * 16 + n] = (__bf16)(o[j][i] * inv);
  }
}

// ---------------------------------------------------------------------------
// Y = gamma*(s-mean)/(std_unbiased+eps)+beta, s = A+R. Optional bf16 copy Yb.
// ---------------------------------------------------------------------------
__global__ __launch_bounds__(256)
void add_layernorm_kernel(const float* __restrict__ A, const float* __restrict__ R,
                          const float* __restrict__ gamma,
                          const float* __restrict__ beta,
                          float* __restrict__ Y, __bf16* __restrict__ Yb) {
  __shared__ float red[256];
  const int tid = threadIdx.x;
  const size_t base = (size_t)blockIdx.x * HIDc;

  float val[3];
#pragma unroll
  for (int r = 0; r < 3; ++r) {
    const int c = tid + 256 * r;
    val[r] = A[base + c] + R[base + c];
  }
  red[tid] = val[0] + val[1] + val[2];
  __syncthreads();
  for (int off = 128; off > 0; off >>= 1) {
    if (tid < off) red[tid] += red[tid + off];
    __syncthreads();
  }
  const float mean = red[0] * (1.0f / (float)HIDc);
  __syncthreads();

  float sq = 0.0f;
#pragma unroll
  for (int r = 0; r < 3; ++r) {
    const float d = val[r] - mean;
    sq += d * d;
  }
  red[tid] = sq;
  __syncthreads();
  for (int off = 128; off > 0; off >>= 1) {
    if (tid < off) red[tid] += red[tid + off];
    __syncthreads();
  }
  const float var = red[0] * (1.0f / (float)(HIDc - 1));  // ddof=1
  const float inv = 1.0f / (sqrtf(var) + EPSc);           // eps added to std
#pragma unroll
  for (int r = 0; r < 3; ++r) {
    const int c = tid + 256 * r;
    const float y = gamma[c] * (val[r] - mean) * inv + beta[c];
    Y[base + c] = y;
    if (Yb) Yb[base + c] = (__bf16)y;
  }
}

// ---------------------------------------------------------------------------
extern "C" void kernel_launch(void* const* d_in, const int* in_sizes, int n_in,
                              void* d_out, int out_size, void* d_ws, size_t ws_size,
                              hipStream_t stream) {
  (void)in_sizes; (void)n_in; (void)out_size; (void)ws_size;

  const float* x    = (const float*)d_in[0];
  const unsigned char* mask = (const unsigned char*)d_in[1];  // numpy bool
  const float* Wq = (const float*)d_in[2];  const float* bq = (const float*)d_in[3];
  const float* Wk = (const float*)d_in[4];  const float* bk = (const float*)d_in[5];
  const float* Wv = (const float*)d_in[6];  const float* bv = (const float*)d_in[7];
  const float* Wm = (const float*)d_in[8];  const float* bm = (const float*)d_in[9];
  const float* W1 = (const float*)d_in[10]; const float* b1 = (const float*)d_in[11];
  const float* W2 = (const float*)d_in[12]; const float* b2 = (const float*)d_in[13];
  const float* g1 = (const float*)d_in[14]; const float* be1 = (const float*)d_in[15];
  const float* g2 = (const float*)d_in[16]; const float* be2 = (const float*)d_in[17];
  float* outp = (float*)d_out;

  const size_t TH  = (size_t)Tc * HIDc;
  const size_t TF  = (size_t)Tc * FFc;
  const size_t Whh = (size_t)HIDc * HIDc;
  const size_t Wfh = (size_t)FFc * HIDc;
  const size_t BS  = (size_t)Bc * Sc;

  // Workspace carve (256B-aligned; total ~190 MB).
  char* wsp = (char*)d_ws;
  size_t off = 0;
  auto carve = [&](size_t bytes) -> void* {
    void* p = wsp + off;
    off += (bytes + 255) & ~(size_t)255;
    return p;
  };
  float*  proj = (float*)carve(TH * 4);
  float*  x1f  = (float*)carve(TH * 4);
  float*  amsk = (float*)carve(BS * 4);
  __bf16* xb   = (__bf16*)carve(TH * 2);
  __bf16* qb   = (__bf16*)carve(TH * 2);
  __bf16* kb   = (__bf16*)carve(TH * 2);
  __bf16* vb   = (__bf16*)carve(TH * 2);
  __bf16* attb = (__bf16*)carve(TH * 2);
  __bf16* x1b  = (__bf16*)carve(TH * 2);
  __bf16* hidb = (__bf16*)carve(TF * 2);
  __bf16* wqb  = (__bf16*)carve(Whh * 2);
  __bf16* wkb  = (__bf16*)carve(Whh * 2);
  __bf16* wvb  = (__bf16*)carve(Whh * 2);
  __bf16* wmb  = (__bf16*)carve(Whh * 2);
  __bf16* w1b  = (__bf16*)carve(Wfh * 2);
  __bf16* w2b  = (__bf16*)carve(Wfh * 2);
  float*  ffn2 = proj;  // proj consumed by LN1 before FFN2 runs

  const dim3 blk(256);
  const dim3 gHID(Tc / 256, HIDc / 64);  // (32, 12)
  const dim3 gFF (Tc / 256, FFc  / 64);  // (32, 48)

  // One-time conversions.
  cvt_f32_to_bf16<<<TH  / 1024, blk, 0, stream>>>(x,  xb);
  cvt_f32_to_bf16<<<Whh / 1024, blk, 0, stream>>>(Wq, wqb);
  cvt_f32_to_bf16<<<Whh / 1024, blk, 0, stream>>>(Wk, wkb);
  cvt_f32_to_bf16<<<Whh / 1024, blk, 0, stream>>>(Wv, wvb);
  cvt_f32_to_bf16<<<Whh / 1024, blk, 0, stream>>>(Wm, wmb);
  cvt_f32_to_bf16<<<Wfh / 1024, blk, 0, stream>>>(W1, w1b);
  cvt_f32_to_bf16<<<Wfh / 1024, blk, 0, stream>>>(W2, w2b);
  cvt_mask_kernel<<<BS / 256,  blk, 0, stream>>>(mask, amsk);

  // Q, K, V projections (bf16 out).
  gemm_bf16_kernel<__bf16, HIDc, HIDc, false><<<gHID, blk, 0, stream>>>(xb, wqb, bq, qb);
  gemm_bf16_kernel<__bf16, HIDc, HIDc, false><<<gHID, blk, 0, stream>>>(xb, wkb, bk, kb);
  gemm_bf16_kernel<__bf16, HIDc, HIDc, false><<<gHID, blk, 0, stream>>>(xb, wvb, bv, vb);

  // Flash attention.
  flash_attn_kernel<<<dim3(Sc / 128, NHc, Bc), blk, 0, stream>>>(qb, kb, vb, amsk, attb);

  // Output projection (fp32) + residual LN1 (fp32 + bf16 x1).
  gemm_bf16_kernel<float, HIDc, HIDc, false><<<gHID, blk, 0, stream>>>(attb, wmb, bm, proj);
  add_layernorm_kernel<<<Tc, blk, 0, stream>>>(x, proj, g1, be1, x1f, x1b);

  // FFN: ReLU(x1 W1^T + b1) (bf16) -> W2^T + b2 (fp32) -> residual LN2 -> out.
  gemm_bf16_kernel<__bf16, FFc, HIDc, true ><<<gFF,  blk, 0, stream>>>(x1b, w1b, b1, hidb);
  gemm_bf16_kernel<float, HIDc, FFc, false><<<gHID, blk, 0, stream>>>(hidb, w2b, b2, ffn2);
  add_layernorm_kernel<<<Tc, blk, 0, stream>>>(x1f, ffn2, g2, be2, outp, nullptr);
}
